// GQA_42047729828019
// MI455X (gfx1250) — matrix operand order repaired and checked by reading, not compiled, not run on hardware
//
#include <hip/hip_runtime.h>

// ---------------------------------------------------------------------------
// GQA attention for MI455X (gfx1250): bf16 WMMA everywhere, f32 accumulate.
// Async LDS staging (GLOBAL_LOAD_ASYNC_TO_LDS_B128) where layout permits.
// ---------------------------------------------------------------------------

typedef __bf16 v16bf  __attribute__((ext_vector_type(16)));
typedef float  v8f    __attribute__((ext_vector_type(8)));
typedef __bf16 bf16x4 __attribute__((ext_vector_type(4)));
typedef __bf16 bf16x2 __attribute__((ext_vector_type(2)));
typedef int    v4i    __attribute__((ext_vector_type(4)));

#define WMMA_BF16(A_, B_, C_) \
  __builtin_amdgcn_wmma_f32_16x16x32_bf16(false, (A_), false, (B_), (short)0, (C_), false, false)

#if defined(__has_builtin)
#  if __has_builtin(__builtin_amdgcn_global_load_async_to_lds_b128) && \
      __has_builtin(__builtin_amdgcn_s_wait_asynccnt)
#    define USE_ASYNC_LDS 1
#  endif
#endif
#ifndef USE_ASYNC_LDS
#  define USE_ASYNC_LDS 0
#endif

#if USE_ASYNC_LDS
typedef __attribute__((address_space(1))) v4i* gptr_v4i;
typedef __attribute__((address_space(3))) v4i* lptr_v4i;
__device__ __forceinline__ void async_copy16(const __bf16* g, __bf16* l) {
  __builtin_amdgcn_global_load_async_to_lds_b128((gptr_v4i)g, (lptr_v4i)l, 0, 0);
}
#endif

constexpr int D_MODEL      = 2048;
constexpr int NUM_HEADS    = 32;
constexpr int NUM_KV_HEADS = 8;
constexpr int D_K          = 64;
constexpr int KV_DIM       = NUM_KV_HEADS * D_K;   // 512
constexpr int BATCH        = 2;
constexpr int SEQ          = 2048;
constexpr int REP          = NUM_HEADS / NUM_KV_HEADS; // 4

// ---------------------------------------------------------------------------
// f32 -> bf16 conversion (vectorized, bandwidth-bound, runs once per buffer)
// ---------------------------------------------------------------------------
__global__ __launch_bounds__(256)
void cvt_f32_bf16_kernel(const float4* __restrict__ src, bf16x4* __restrict__ dst, int n4) {
  int i = blockIdx.x * 256 + threadIdx.x;
  if (i < n4) {
    float4 f = src[i];
    bf16x4 o;
    o[0] = (__bf16)f.x; o[1] = (__bf16)f.y; o[2] = (__bf16)f.z; o[3] = (__bf16)f.w;
    dst[i] = o;
  }
}

// ---------------------------------------------------------------------------
// Tiled WMMA GEMM:  C[M,N] = A[M,K](bf16) * B[K,N](bf16) + bias[N](f32)
// Block: 256 threads = 8 waves, macro tile 128(M) x 128(N), K-step 32.
// Wave grid 4(M) x 2(N); each wave computes a 32x64 sub-tile (2x4 WMMAs),
// so every LDS fragment gather feeds 2-4 WMMAs.
// ---------------------------------------------------------------------------
template<bool OUT_F32>
__global__ __launch_bounds__(256)
void gemm_bias_kernel(const __bf16* __restrict__ A, const __bf16* __restrict__ B,
                      const float* __restrict__ bias,
                      float* __restrict__ outF, __bf16* __restrict__ outB,
                      int M, int N, int K)
{
  __shared__ __bf16 Asm[128][40];  // 80B row stride: 16B-aligned for b128 async,
                                   // bank step 20 dwords -> conflict-free gathers
  __shared__ __bf16 Bt[128][36];   // B tile stored transposed: Bt[n][k]

  const int tid  = threadIdx.x;
  const int lane = tid & 31;
  const int half = lane >> 4;
  const int l16  = lane & 15;
  const int wave = tid >> 5;
  const int wm   = wave & 3;   // 0..3
  const int wn   = wave >> 2;  // 0..1
  const int rowBase = blockIdx.y * 128;
  const int colBase = blockIdx.x * 128;

  v8f acc[2][4];
  #pragma unroll
  for (int i = 0; i < 2; ++i)
    #pragma unroll
    for (int j = 0; j < 4; ++j)
      #pragma unroll
      for (int r = 0; r < 8; ++r) acc[i][j][r] = 0.0f;

  for (int kt = 0; kt < K; kt += 32) {
    // ---- stage A tile 128x32 (row-major) ----
#if USE_ASYNC_LDS
    #pragma unroll
    for (int i = 0; i < 2; ++i) {
      int c  = tid + i * 256;
      int r  = c >> 2;            // 4 chunks of 8 elems (16B) per row
      int kc = (c & 3) << 3;
      async_copy16(A + (size_t)(rowBase + r) * K + kt + kc, &Asm[r][kc]);
    }
#else
    #pragma unroll
    for (int i = 0; i < 4; ++i) {
      int c  = tid + i * 256;
      int r  = c >> 3;            // 8 chunks of 4 per row
      int kc = (c & 7) << 2;
      *(bf16x4*)&Asm[r][kc] =
          *(const bf16x4*)(A + (size_t)(rowBase + r) * K + kt + kc);
    }
#endif
    // ---- stage B tile 32x128, transposed into Bt[n][k] ----
    #pragma unroll
    for (int i = 0; i < 4; ++i) {
      int c  = tid + i * 256;
      int r  = c >> 5;            // k index 0..31 (32 chunks per row)
      int nc = (c & 31) << 2;
      bf16x4 vl = *(const bf16x4*)(B + (size_t)(kt + r) * N + colBase + nc);
      Bt[nc + 0][r] = vl[0]; Bt[nc + 1][r] = vl[1];
      Bt[nc + 2][r] = vl[2]; Bt[nc + 3][r] = vl[3];
    }
#if USE_ASYNC_LDS
    __builtin_amdgcn_s_wait_asynccnt(0);
#endif
    __syncthreads();

    // ---- build fragments (ISA 7.12.2 layouts) ----
    v16bf af[2], bfr[4];
    #pragma unroll
    for (int i = 0; i < 2; ++i) {
      int rowA = wm * 32 + i * 16 + l16;
      #pragma unroll
      for (int v = 0; v < 8; ++v) {
        int kb = ((v & 3) << 1) + ((v >> 2) << 4) + (half << 3);
        bf16x2 p = *(const bf16x2*)&Asm[rowA][kb];
        af[i][2 * v] = p[0]; af[i][2 * v + 1] = p[1];
      }
    }
    #pragma unroll
    for (int j = 0; j < 4; ++j) {
      int colB = wn * 64 + j * 16 + l16;
      #pragma unroll
      for (int v = 0; v < 8; ++v) {
        int kb = ((v & 3) << 1) + ((v >> 2) << 4) + (half << 3);
        bf16x2 p = *(const bf16x2*)&Bt[colB][kb];
        bfr[j][2 * v] = p[0]; bfr[j][2 * v + 1] = p[1];
      }
    }
    #pragma unroll
    for (int i = 0; i < 2; ++i)
      #pragma unroll
      for (int j = 0; j < 4; ++j)
        acc[i][j] = WMMA_BF16(af[i], bfr[j], acc[i][j]);
    __syncthreads();
  }

  // ---- epilogue: C layout row = r + 8*half, col = l16 ----
  #pragma unroll
  for (int i = 0; i < 2; ++i)
    #pragma unroll
    for (int j = 0; j < 4; ++j) {
      int   colg = colBase + wn * 64 + j * 16 + l16;
      float bv   = bias[colg];
      #pragma unroll
      for (int r = 0; r < 8; ++r) {
        int   rowg = rowBase + wm * 32 + i * 16 + r + (half << 3);
        float v    = acc[i][j][r] + bv;
        if constexpr (OUT_F32) outF[(size_t)rowg * N + colg] = v;
        else                   outB[(size_t)rowg * N + colg] = (__bf16)v;
      }
    }
}

// ---------------------------------------------------------------------------
// Flash attention (non-causal), GQA: head h uses kv head h/REP.
// Block: 128 threads = 4 waves; each wave owns 16 query rows; 64 q rows/block.
// Iterates keys in tiles of 32; QK^T and PV via bf16 WMMA, f32 online softmax.
// K tile staged via async-to-LDS; V tile transposed via VGPR path.
// ---------------------------------------------------------------------------
__global__ __launch_bounds__(128)
void gqa_attention_kernel(const __bf16* __restrict__ Qg, const __bf16* __restrict__ Kg,
                          const __bf16* __restrict__ Vg, __bf16* __restrict__ ctx)
{
  __shared__ __bf16 Qs[64][68];      // 64 q rows x 64 d
  __shared__ __bf16 Ks[32][72];      // 144B stride: 16B-aligned, conflict-free
  __shared__ __bf16 Vt[64][36];      // V transposed: Vt[d][key]
  __shared__ __bf16 Ps[4][16][36];   // per-wave P tile 16 x 32

  const int tid  = threadIdx.x;
  const int lane = tid & 31;
  const int half = lane >> 4;
  const int l16  = lane & 15;
  const int wave = tid >> 5;

  const int bh = blockIdx.y;
  const int b  = bh / NUM_HEADS;
  const int h  = bh % NUM_HEADS;
  const int g  = h / REP;
  const int qbase = blockIdx.x * 64;
  const size_t qrow0 = (size_t)b * SEQ + qbase;
  const size_t krow0 = (size_t)b * SEQ;

  // ---- stage Q tile 64x64 ----
  #pragma unroll
  for (int i = 0; i < 8; ++i) {
    int c  = tid + i * 128;
    int r  = c >> 4;
    int dc = (c & 15) << 2;
    *(bf16x4*)&Qs[r][dc] =
        *(const bf16x4*)(Qg + (qrow0 + r) * D_MODEL + h * D_K + dc);
  }
  __syncthreads();

  // ---- Q A-fragments (kept in registers for the whole kernel) ----
  v16bf qf[2];
  {
    int rowA = wave * 16 + l16;
    #pragma unroll
    for (int ks = 0; ks < 2; ++ks)
      #pragma unroll
      for (int v = 0; v < 8; ++v) {
        int kb = ks * 32 + ((v & 3) << 1) + ((v >> 2) << 4) + (half << 3);
        bf16x2 p = *(const bf16x2*)&Qs[rowA][kb];
        qf[ks][2 * v] = p[0]; qf[ks][2 * v + 1] = p[1];
      }
  }

  float mrun[8], lrun[8];
  v8f   o[4];
  #pragma unroll
  for (int r = 0; r < 8; ++r) { mrun[r] = -1.0e30f; lrun[r] = 0.0f; }
  #pragma unroll
  for (int dt = 0; dt < 4; ++dt)
    #pragma unroll
    for (int r = 0; r < 8; ++r) o[dt][r] = 0.0f;

  for (int kt = 0; kt < SEQ; kt += 32) {
    __syncthreads();   // previous iteration's Ks/Vt readers done
    // ---- stage K tile 32x64 (async, row-major) ----
#if USE_ASYNC_LDS
    #pragma unroll
    for (int i = 0; i < 2; ++i) {
      int c  = tid + i * 128;
      int r  = c >> 3;            // 8 chunks of 8 elems (16B) per row
      int dc = (c & 7) << 3;
      async_copy16(Kg + (krow0 + kt + r) * KV_DIM + g * D_K + dc, &Ks[r][dc]);
    }
#endif
    // ---- stage V tile 32x64 transposed (and K fallback), prefetch next ----
    #pragma unroll
    for (int i = 0; i < 4; ++i) {
      int c  = tid + i * 128;
      int r  = c >> 4;
      int dc = (c & 15) << 2;
      const __bf16* vp = Vg + (krow0 + kt + r) * KV_DIM + g * D_K + dc;
#if !USE_ASYNC_LDS
      const __bf16* kp = Kg + (krow0 + kt + r) * KV_DIM + g * D_K + dc;
      *(bf16x4*)&Ks[r][dc] = *(const bf16x4*)kp;
#endif
      bf16x4 vv = *(const bf16x4*)vp;
      Vt[dc + 0][r] = vv[0]; Vt[dc + 1][r] = vv[1];
      Vt[dc + 2][r] = vv[2]; Vt[dc + 3][r] = vv[3];
      if (kt + 32 < SEQ) __builtin_prefetch(vp + 32 * KV_DIM, 0, 0);
    }
#if USE_ASYNC_LDS
    __builtin_amdgcn_s_wait_asynccnt(0);
#endif
    __syncthreads();

    // ---- scores: S[16 x 32] = Q(16x64) * K^T(64x32), 4 WMMAs ----
    v8f sc[2];
    #pragma unroll
    for (int t = 0; t < 2; ++t)
      #pragma unroll
      for (int r = 0; r < 8; ++r) sc[t][r] = 0.0f;

    #pragma unroll
    for (int tn = 0; tn < 2; ++tn) {
      int key = tn * 16 + l16;   // B-matrix column = key
      #pragma unroll
      for (int ks = 0; ks < 2; ++ks) {
        v16bf kf;
        #pragma unroll
        for (int v = 0; v < 8; ++v) {
          int db = ks * 32 + ((v & 3) << 1) + ((v >> 2) << 4) + (half << 3);   // B k-index = d
          bf16x2 p = *(const bf16x2*)&Ks[key][db];
          kf[2 * v] = p[0]; kf[2 * v + 1] = p[1];
        }
        sc[tn] = WMMA_BF16(qf[ks], kf, sc[tn]);
      }
    }

    // ---- online softmax over this 32-key tile ----
    #pragma unroll
    for (int r = 0; r < 8; ++r) {
      float s0 = sc[0][r] * 0.125f;     // 1/sqrt(64)
      float s1 = sc[1][r] * 0.125f;
      float tmax = fmaxf(s0, s1);       // reduce across the 16-lane half
      tmax = fmaxf(tmax, __shfl_xor(tmax, 1, 32));
      tmax = fmaxf(tmax, __shfl_xor(tmax, 2, 32));
      tmax = fmaxf(tmax, __shfl_xor(tmax, 4, 32));
      tmax = fmaxf(tmax, __shfl_xor(tmax, 8, 32));
      float mn    = fmaxf(mrun[r], tmax);
      float alpha = __expf(mrun[r] - mn);
      mrun[r] = mn;
      float p0 = __expf(s0 - mn);
      float p1 = __expf(s1 - mn);
      float rs = p0 + p1;
      rs += __shfl_xor(rs, 1, 32);
      rs += __shfl_xor(rs, 2, 32);
      rs += __shfl_xor(rs, 4, 32);
      rs += __shfl_xor(rs, 8, 32);
      lrun[r] = lrun[r] * alpha + rs;
      #pragma unroll
      for (int dt = 0; dt < 4; ++dt) o[dt][r] *= alpha;   // rescale accumulators
      int prow = r + (half << 3);                         // C layout -> LDS
      Ps[wave][prow][l16]      = (__bf16)p0;
      Ps[wave][prow][16 + l16] = (__bf16)p1;
    }

    // ---- relayout P (C->A) and PV: O += P(16x32) * V(32x64), 4 WMMAs ----
    v16bf pf;
    #pragma unroll
    for (int v = 0; v < 8; ++v) {
      int kb = ((v & 3) << 1) + ((v >> 2) << 4) + (half << 3);
      bf16x2 p = *(const bf16x2*)&Ps[wave][l16][kb];
      pf[2 * v] = p[0]; pf[2 * v + 1] = p[1];
    }
    #pragma unroll
    for (int dt = 0; dt < 4; ++dt) {
      int dcol = dt * 16 + l16;     // B-matrix column = d
      v16bf vf;
      #pragma unroll
      for (int v = 0; v < 8; ++v) {
        int kb = ((v & 3) << 1) + ((v >> 2) << 4) + (half << 3);   // B k-index = key
        bf16x2 p = *(const bf16x2*)&Vt[dcol][kb];
        vf[2 * v] = p[0]; vf[2 * v + 1] = p[1];
      }
      o[dt] = WMMA_BF16(pf, vf, o[dt]);
    }
  }

  // ---- normalize and write ctx (bf16, [b, s, h, d] layout == [M, D_MODEL]) ----
  #pragma unroll
  for (int r = 0; r < 8; ++r) {
    float inv  = 1.0f / lrun[r];
    int   rowg = qbase + wave * 16 + r + (half << 3);
    size_t base = ((size_t)b * SEQ + rowg) * D_MODEL + h * D_K;
    #pragma unroll
    for (int dt = 0; dt < 4; ++dt)
      ctx[base + dt * 16 + l16] = (__bf16)(o[dt][r] * inv);
  }
}

// ---------------------------------------------------------------------------
// Host-side orchestration
// ---------------------------------------------------------------------------
extern "C" void kernel_launch(void* const* d_in, const int* in_sizes, int n_in,
                              void* d_out, int out_size, void* d_ws, size_t ws_size,
                              hipStream_t stream) {
  const float* x  = (const float*)d_in[0];
  const float* Wq = (const float*)d_in[1];
  const float* bq = (const float*)d_in[2];
  const float* Wk = (const float*)d_in[3];
  const float* bk = (const float*)d_in[4];
  const float* Wv = (const float*)d_in[5];
  const float* bv = (const float*)d_in[6];
  const float* Wo = (const float*)d_in[7];
  const float* bo = (const float*)d_in[8];
  float* out = (float*)d_out;

  constexpr size_t MB = 1ull << 20;
  char* ws = (char*)d_ws;
  __bf16* xb  = (__bf16*)(ws +  0 * MB);  // 16 MB : x bf16       [4096 x 2048]
  __bf16* Wqb = (__bf16*)(ws + 16 * MB);  //  8 MB : Wq bf16      [2048 x 2048]
  __bf16* Wkb = (__bf16*)(ws + 24 * MB);  //  2 MB : Wk bf16      [2048 x  512]
  __bf16* Wvb = (__bf16*)(ws + 26 * MB);  //  2 MB : Wv bf16
  __bf16* Wob = (__bf16*)(ws + 28 * MB);  //  8 MB : Wo bf16
  __bf16* Qb  = (__bf16*)(ws + 36 * MB);  // 16 MB : Q bf16       [4096 x 2048]
  __bf16* Kb  = (__bf16*)(ws + 52 * MB);  //  4 MB : K bf16       [4096 x  512]
  __bf16* Vb  = (__bf16*)(ws + 56 * MB);  //  4 MB : V bf16
  __bf16* Cb  = (__bf16*)(ws + 60 * MB);  // 16 MB : ctx bf16     [4096 x 2048]

  const size_t M = (size_t)BATCH * SEQ;   // 4096

  auto cvt = [&](const float* s, __bf16* d, size_t n) {
    int n4 = (int)(n / 4);
    cvt_f32_bf16_kernel<<<(n4 + 255) / 256, 256, 0, stream>>>(
        (const float4*)s, (bf16x4*)d, n4);
  };
  cvt(x,  xb,  M * D_MODEL);
  cvt(Wq, Wqb, (size_t)D_MODEL * D_MODEL);
  cvt(Wk, Wkb, (size_t)D_MODEL * KV_DIM);
  cvt(Wv, Wvb, (size_t)D_MODEL * KV_DIM);
  cvt(Wo, Wob, (size_t)D_MODEL * D_MODEL);

  dim3 blk(256);
  // Q/K/V projections (bf16 out)
  gemm_bias_kernel<false><<<dim3(D_MODEL / 128, M / 128), blk, 0, stream>>>(
      xb, Wqb, bq, nullptr, Qb, (int)M, D_MODEL, D_MODEL);
  gemm_bias_kernel<false><<<dim3(KV_DIM / 128, M / 128), blk, 0, stream>>>(
      xb, Wkb, bk, nullptr, Kb, (int)M, KV_DIM, D_MODEL);
  gemm_bias_kernel<false><<<dim3(KV_DIM / 128, M / 128), blk, 0, stream>>>(
      xb, Wvb, bv, nullptr, Vb, (int)M, KV_DIM, D_MODEL);

  // flash attention (one block per 64 q rows per (b,h))
  gqa_attention_kernel<<<dim3(SEQ / 64, BATCH * NUM_HEADS), dim3(128), 0, stream>>>(
      Qb, Kb, Vb, Cb);

  // output projection (f32 out -> d_out)
  gemm_bias_kernel<true><<<dim3(D_MODEL / 128, M / 128), blk, 0, stream>>>(
      Cb, Wob, bo, out, nullptr, (int)M, D_MODEL, D_MODEL);
}
